// RetinaNetHead_8048768713007
// MI455X (gfx1250) — matrix operand order repaired and checked by reading, not compiled, not run on hardware
//
#include <hip/hip_runtime.h>
#include <cstdint>
#include <cstddef>

typedef __attribute__((ext_vector_type(16))) _Float16 v16h;
typedef __attribute__((ext_vector_type(8)))  float    v8f;
typedef unsigned int v4u __attribute__((ext_vector_type(4)));
typedef int          v4i __attribute__((ext_vector_type(4)));
typedef int          v8i __attribute__((ext_vector_type(8)));

#define IN_CH        256
#define NUM_CLASSES  15
#define NUM_ANCHORS  4
#define NTOT         87296                      // total anchors per image across levels
#define SCORES_TOTAL (8 * NTOT * NUM_CLASSES)   // floats before boxes section

// ---------------------------------------------------------------------------
// Tensor Data Mover: load one contiguous 16KB weight chunk into LDS.
// D# per CDNA5 ISA 8.3/8.4. Toolchain arity dispatched via TDM header probe.
// ---------------------------------------------------------------------------
__device__ __forceinline__ void tdm_load_chunk(unsigned long long ga,
                                               unsigned ldsOff) {
  v4u g0;
  g0.x = 1u;                                   // count=1, user descriptor
  g0.y = ldsOff;                               // lds_addr (bytes)
  g0.z = (unsigned)(ga & 0xffffffffull);       // global_addr[31:0]
  g0.w = (unsigned)((ga >> 32) & 0x1ffffffull) // global_addr[56:32]
         | (2u << 30);                         // type = 2 (image)
  const unsigned TD0 = 2304u * 256u;           // tensor_dim0 (elements)
  v8i g1;
  g1[0] = (int)(1u << 16);                     // data_size = 1 -> 2 bytes
  g1[1] = (int)((TD0 & 0xffffu) << 16);        // tensor_dim0[15:0] @ bits 63:48
  g1[2] = (int)(((TD0 >> 16) & 0xffffu)        // tensor_dim0[31:16]
         | (1u << 16));                        // tensor_dim1[15:0] = 1
  g1[3] = (int)(8192u << 16);                  // tensor_dim1 hi=0; tile_dim0=8192
  g1[4] = (int)1;                              // tile_dim1=1, tile_dim2=0
  g1[5] = (int)TD0;                            // tensor_dim0_stride[31:0]
  g1[6] = 0;                                   // stride0 hi; stride1 lo = 0
  g1[7] = 0;                                   // stride1 hi
  v4i g2; g2[0] = 1; g2[1] = 1; g2[2] = 0; g2[3] = 0;  // dim2=1, dim3=1
  v4i g3; g3[0] = 0; g3[1] = 0; g3[2] = 0; g3[3] = 0;
#if __has_include(<hip/amd_detail/amd_gfx1250_TDM.h>)
  v8i gz; gz[0]=0; gz[1]=0; gz[2]=0; gz[3]=0; gz[4]=0; gz[5]=0; gz[6]=0; gz[7]=0;
  __builtin_amdgcn_tensor_load_to_lds(g0, g1, g2, g3, gz, 0);
#else
  __builtin_amdgcn_tensor_load_to_lds(g0, g1, g2, g3, 0);
#endif
}

// ---------------------------------------------------------------------------
// Zero fill (16B granules, grid-stride)
// ---------------------------------------------------------------------------
__global__ void fill_zero16(uint4* __restrict__ p, long n16) {
  long i = blockIdx.x * (long)blockDim.x + threadIdx.x;
  long stride = (long)gridDim.x * blockDim.x;
  uint4 z; z.x = 0u; z.y = 0u; z.z = 0u; z.w = 0u;
  for (; i < n16; i += stride) p[i] = z;
}

// ---------------------------------------------------------------------------
// NCHW f32 -> padded NHWC f16 (pad = 1 ring of zeros, pre-zeroed by fill)
// ---------------------------------------------------------------------------
__global__ void nchw_to_padded_nhwc(const float* __restrict__ src,
                                    _Float16* __restrict__ dst,
                                    int H, int W) {
  long total = (long)8 * IN_CH * H * W;
  long i = blockIdx.x * (long)blockDim.x + threadIdx.x;
  if (i >= total) return;
  int x = (int)(i % W); long t = i / W;
  int y = (int)(t % H); t /= H;
  int c = (int)(t % IN_CH); int b = (int)(t / IN_CH);
  long o = (((long)b * (H + 2) + (y + 1)) * (W + 2) + (x + 1)) * IN_CH + c;
  dst[o] = (_Float16)src[i];
}

// ---------------------------------------------------------------------------
// Pack [OC=256][IC=256][3][3] f32 weights into WMMA-A fragment order (f16).
// Fragment = (kpos, icChunk, mtile): 32 lanes x 16 halves, lane-major.
//   lane 0..15 : M = lane,    K runs [0..15]  (+32*icc)
//   lane 16..31: M = lane-16, K runs [16..31] (+32*icc)
// One (kpos,icc) slab = 16 mtiles * 1KB = 16KB contiguous (TDM chunk).
// ---------------------------------------------------------------------------
__global__ void pack_conv3x3_weights(const float* __restrict__ w,
                                     _Float16* __restrict__ out) {
  int idx = blockIdx.x * blockDim.x + threadIdx.x;
  if (idx >= 2304 * 256) return;
  int t     = idx & 15;
  int lane  = (idx >> 4) & 31;
  int frag  = idx >> 9;
  int mtile = frag & 15;        // 16 M tiles of 16 OC
  int icc   = (frag >> 4) & 7;  // 8 chunks of 32 IC
  int kpos  = frag >> 7;        // 0..8 -> (ky,kx)
  int m     = lane & 15;
  int grp   = lane >> 4;
  int run   = t >> 3;
  int off   = t & 7;
  int ic    = icc * 32 + grp * 16 + run * 8 + off;
  int oc    = mtile * 16 + m;
  int ky    = kpos / 3, kx = kpos - ky * 3;
  out[idx] = (_Float16)w[(((long)oc * 256 + ic) * 3 + ky) * 3 + kx];
}

// ---------------------------------------------------------------------------
// 3x3 conv, IC=OC=256, implicit GEMM on the WMMA pipe.
// Block = 256 threads = 8 waves arranged 4(M) x 2(N).
// Each wave: 4 M-tiles x 2 N-tiles of 16x16 (f32 accum), K = 9*256 = 2304.
// Weights: TDM double-buffered global->LDS (16KB chunks), A-frags via ds reads.
// Activations: direct global b128 (halo reuse lives in WGP$/L2).
// In/out: padded NHWC f16; fused bias + ReLU.
// ---------------------------------------------------------------------------
__global__ __launch_bounds__(256) void conv3x3_wmma_relu(
    const _Float16* __restrict__ in, const _Float16* __restrict__ wpk,
    const float* __restrict__ bias, _Float16* __restrict__ out,
    int H, int W) {
  __shared__ __align__(64) _Float16 lds_w[2][8192];   // 2 x 16KB chunks

  const int lane  = threadIdx.x & 31;
  const int wave  = threadIdx.x >> 5;
  const int waveM = wave & 3;   // M base = waveM*64
  const int waveN = wave >> 2;  // N base += waveN*32
  const int b     = blockIdx.z;
  const int Wp    = W + 2;
  const int Nbase = blockIdx.x * 64 + waveN * 32;
  const int l15   = lane & 15;
  const int icOff = (lane >> 4) << 4;   // B frag: hi lanes take K 16..31

  int  py[2], px[2];
  long pixBase[2];
  #pragma unroll
  for (int nt = 0; nt < 2; ++nt) {
    int p = Nbase + nt * 16 + l15;
    int y = p / W, x = p - y * W;
    py[nt] = y; px[nt] = x;
    // padded coords: input pixel (y+ky-1 +1, x+kx-1 +1) = (y+ky, x+kx)
    pixBase[nt] = (((long)b * (H + 2) + y) * Wp + x) * IN_CH;
  }

  v8f acc[4][2];
  v8f zero = {};
  #pragma unroll
  for (int mt = 0; mt < 4; ++mt)
    #pragma unroll
    for (int nt = 0; nt < 2; ++nt) acc[mt][nt] = zero;

  // uniform (SGPR) values for the TDM path
  const bool isW0 = (__builtin_amdgcn_readfirstlane((int)threadIdx.x) < 32);
  const unsigned long long wbase =
      (unsigned long long)(uintptr_t)wpk;
  const unsigned lds0 = (unsigned)__builtin_amdgcn_readfirstlane(
      (int)(uintptr_t)(void*)&lds_w[0][0]);
  const unsigned lds1 = (unsigned)__builtin_amdgcn_readfirstlane(
      (int)(uintptr_t)(void*)&lds_w[1][0]);

  if (isW0) tdm_load_chunk(wbase, lds0);              // prologue: chunk 0

  #pragma unroll 1
  for (int chunk = 0; chunk < 72; ++chunk) {
    if (isW0) __builtin_amdgcn_s_wait_tensorcnt(0);   // this chunk landed
    __syncthreads();                                  // publish buffer
    if (isW0 && chunk < 71) {                         // kick next DMA
      unsigned long long ga =
          wbase + (unsigned long long)(unsigned)__builtin_amdgcn_readfirstlane(
                      (chunk + 1) * 16384);
      tdm_load_chunk(ga, ((chunk + 1) & 1) ? lds1 : lds0);
    }

    const int kpos = chunk >> 3, icc = chunk & 7;
    const int ky = kpos / 3, kx = kpos - ky * 3;
    const long ro0 = pixBase[0] + ((long)ky * Wp + kx) * IN_CH + icc * 32 + icOff;
    const long ro1 = pixBase[1] + ((long)ky * Wp + kx) * IN_CH + icc * 32 + icOff;

    v16h b0 = *(const v16h*)(in + ro0);
    v16h b1 = *(const v16h*)(in + ro1);
    __builtin_prefetch(in + ro0 + 64, 0, 1);          // stream next IC chunk

    const _Float16* lw = &lds_w[chunk & 1][0] + (waveM * 4) * 512 + lane * 16;
    v16h a0 = *(const v16h*)(lw + 0 * 512);
    v16h a1 = *(const v16h*)(lw + 1 * 512);
    v16h a2 = *(const v16h*)(lw + 2 * 512);
    v16h a3 = *(const v16h*)(lw + 3 * 512);

    acc[0][0] = __builtin_amdgcn_wmma_f32_16x16x32_f16(false, a0, false, b0, (short)0, acc[0][0], false, false);
    acc[0][1] = __builtin_amdgcn_wmma_f32_16x16x32_f16(false, a0, false, b1, (short)0, acc[0][1], false, false);
    acc[1][0] = __builtin_amdgcn_wmma_f32_16x16x32_f16(false, a1, false, b0, (short)0, acc[1][0], false, false);
    acc[1][1] = __builtin_amdgcn_wmma_f32_16x16x32_f16(false, a1, false, b1, (short)0, acc[1][1], false, false);
    acc[2][0] = __builtin_amdgcn_wmma_f32_16x16x32_f16(false, a2, false, b0, (short)0, acc[2][0], false, false);
    acc[2][1] = __builtin_amdgcn_wmma_f32_16x16x32_f16(false, a2, false, b1, (short)0, acc[2][1], false, false);
    acc[3][0] = __builtin_amdgcn_wmma_f32_16x16x32_f16(false, a3, false, b0, (short)0, acc[3][0], false, false);
    acc[3][1] = __builtin_amdgcn_wmma_f32_16x16x32_f16(false, a3, false, b1, (short)0, acc[3][1], false, false);
  }

  // C/D layout: VGPR i -> M = i (lanes 0..15) / i+8 (lanes 16..31), N = lane&15
  const int ocLaneAdd = (lane >> 4) << 3;
  #pragma unroll
  for (int nt = 0; nt < 2; ++nt) {
    long obase = (((long)b * (H + 2) + (py[nt] + 1)) * Wp + (px[nt] + 1)) * IN_CH;
    #pragma unroll
    for (int mt = 0; mt < 4; ++mt) {
      int ocBase = waveM * 64 + mt * 16 + ocLaneAdd;
      #pragma unroll
      for (int i = 0; i < 8; ++i) {
        int oc = ocBase + i;
        float v = acc[mt][nt][i] + bias[oc];
        v = v > 0.f ? v : 0.f;
        out[obase + oc] = (_Float16)v;
      }
    }
  }
}

// ---------------------------------------------------------------------------
// Heads (1x1 convs, <1% of FLOPs) + anchor gen + rbox decode + scores.
// One thread per (b, y, x, anchor). Head weights staged in LDS as f16.
// Rows in LDS: 0..19 reg_head (a*5+j), 20..23 conf (a), 24..83 cls (a*15+k).
// ---------------------------------------------------------------------------
__global__ __launch_bounds__(256) void head_epilogue(
    const _Float16* __restrict__ rbuf, const _Float16* __restrict__ cbuf,
    const float* __restrict__ reg_head_w, const float* __restrict__ reg_head_b,
    const float* __restrict__ cls_head_w, const float* __restrict__ cls_head_b,
    const float* __restrict__ conf_w, const float* __restrict__ conf_b,
    float* __restrict__ outbuf, int H, int W, float strideF, int lvlOff) {
  __shared__ _Float16 wl[84 * 256];
  for (int k = threadIdx.x; k < 84 * 256; k += blockDim.x) {
    int row = k >> 8, ch = k & 255;
    float v;
    if (row < 20)      v = reg_head_w[row * 256 + ch];
    else if (row < 24) v = conf_w[(row - 20) * 256 + ch];
    else               v = cls_head_w[(row - 24) * 256 + ch];
    wl[k] = (_Float16)v;
  }
  __syncthreads();

  long gid = blockIdx.x * (long)blockDim.x + threadIdx.x;
  long total = (long)8 * H * W * NUM_ANCHORS;
  if (gid >= total) return;
  int a = (int)(gid & 3);
  long loc = gid >> 2;
  int x = (int)(loc % W); long t2 = loc / W;
  int y = (int)(t2 % H); int b = (int)(t2 / H);

  long pb = (((long)b * (H + 2) + (y + 1)) * (W + 2) + (x + 1)) * IN_CH;
  const _Float16* r = rbuf + pb;
  const _Float16* c = cbuf + pb;

  float bb[5]  = {0.f, 0.f, 0.f, 0.f, 0.f};
  float cf     = 0.f;
  float cls[NUM_CLASSES];
  #pragma unroll
  for (int k = 0; k < NUM_CLASSES; ++k) cls[k] = 0.f;

  const _Float16* wb = &wl[(a * 5) * 256];
  const _Float16* wc = &wl[(20 + a) * 256];
  const _Float16* wk = &wl[(24 + a * NUM_CLASSES) * 256];
  for (int ch = 0; ch < 256; ++ch) {
    float rv = (float)r[ch];
    float cv = (float)c[ch];
    #pragma unroll
    for (int j = 0; j < 5; ++j) bb[j] += rv * (float)wb[j * 256 + ch];
    cf += rv * (float)wc[ch];
    #pragma unroll
    for (int k = 0; k < NUM_CLASSES; ++k) cls[k] += cv * (float)wk[k * 256 + ch];
  }
  #pragma unroll
  for (int j = 0; j < 5; ++j) bb[j] += reg_head_b[a * 5 + j];
  cf += conf_b[a];

  // anchor (base_size = stride, scale 4, ratio 6, angle per a)
  const float PI = 3.14159265358979323846f;
  float ctr = 0.5f * (strideF - 1.f);
  float rx = ctr + x * strideF, ry = ctr + y * strideF;
  float rw = strideF * 4.f * 0.4082482904638631f;   // 1/sqrt(6)
  float rh = strideF * 4.f * 2.449489742783178f;    // sqrt(6)
  float ra = (-0.125f + 0.25f * (float)a) * PI;

  const float MAXR = 13.815510557964274f;           // |log(1e-6)|
  float dw = fminf(fmaxf(bb[2], -MAXR), MAXR);
  float dh = fminf(fmaxf(bb[3], -MAXR), MAXR);
  float gx = bb[0] * rw + rx;
  float gy = bb[1] * rh + ry;
  float gw = rw * __expf(dw);
  float gh = rh * __expf(dh);
  float ga = fmodf(bb[4] + ra + 0.25f * PI, PI);
  if (ga < 0.f) ga += PI;
  ga -= 0.25f * PI;

  long n = lvlOff + ((long)y * W + x) * NUM_ANCHORS + a;
  float sc = 1.f / (1.f + __expf(-cf));
  float* scores = outbuf + (long)b * NTOT * NUM_CLASSES + n * NUM_CLASSES;
  #pragma unroll
  for (int k = 0; k < NUM_CLASSES; ++k)
    scores[k] = sc / (1.f + __expf(-(cls[k] + cls_head_b[a * NUM_CLASSES + k])));
  float* boxes = outbuf + SCORES_TOTAL + (long)b * NTOT * 5 + n * 5;
  boxes[0] = gx; boxes[1] = gy; boxes[2] = gw; boxes[3] = gh; boxes[4] = ga;
}

// ---------------------------------------------------------------------------
extern "C" void kernel_launch(void* const* d_in, const int* in_sizes, int n_in,
                              void* d_out, int out_size, void* d_ws, size_t ws_size,
                              hipStream_t stream) {
  (void)in_sizes; (void)n_in; (void)out_size; (void)ws_size;
  const float* feats[5] = {(const float*)d_in[0], (const float*)d_in[1],
                           (const float*)d_in[2], (const float*)d_in[3],
                           (const float*)d_in[4]};
  const float* reg_w0 = (const float*)d_in[5];
  const float* reg_b0 = (const float*)d_in[6];
  const float* reg_w1 = (const float*)d_in[7];
  const float* reg_b1 = (const float*)d_in[8];
  const float* cls_w0 = (const float*)d_in[9];
  const float* cls_b0 = (const float*)d_in[10];
  const float* cls_w1 = (const float*)d_in[11];
  const float* cls_b1 = (const float*)d_in[12];
  const float* reg_head_w = (const float*)d_in[13];
  const float* reg_head_b = (const float*)d_in[14];
  const float* cls_head_w = (const float*)d_in[15];
  const float* cls_head_b = (const float*)d_in[16];
  const float* conf_w = (const float*)d_in[17];
  const float* conf_b = (const float*)d_in[18];
  float* out = (float*)d_out;

  char* ws = (char*)d_ws;
  const size_t PAD_BYTES = (size_t)8 * 130 * 130 * 256 * 2;  // level-0 size, reused
  _Float16* padA = (_Float16*)(ws);
  _Float16* padB = (_Float16*)(ws + PAD_BYTES);
  _Float16* padC = (_Float16*)(ws + 2 * PAD_BYTES);
  const size_t WPK_BYTES = (size_t)2304 * 256 * 2;
  _Float16* wpk0 = (_Float16*)(ws + 3 * PAD_BYTES);
  _Float16* wpk1 = (_Float16*)(ws + 3 * PAD_BYTES + 1 * WPK_BYTES);
  _Float16* wpk2 = (_Float16*)(ws + 3 * PAD_BYTES + 2 * WPK_BYTES);
  _Float16* wpk3 = (_Float16*)(ws + 3 * PAD_BYTES + 3 * WPK_BYTES);

  // pack tower weights into WMMA fragment order
  {
    int n = 2304 * 256;
    int blocks = (n + 255) / 256;
    pack_conv3x3_weights<<<blocks, 256, 0, stream>>>(reg_w0, wpk0);
    pack_conv3x3_weights<<<blocks, 256, 0, stream>>>(reg_w1, wpk1);
    pack_conv3x3_weights<<<blocks, 256, 0, stream>>>(cls_w0, wpk2);
    pack_conv3x3_weights<<<blocks, 256, 0, stream>>>(cls_w1, wpk3);
  }

  const int   Hs[5]      = {128, 64, 32, 16, 8};
  const float strides[5] = {8.f, 16.f, 32.f, 64.f, 128.f};
  const int   lvlOff[5]  = {0, 65536, 81920, 86016, 87040};

  for (int l = 0; l < 5; ++l) {
    const int H = Hs[l], W = Hs[l];
    const long padElems = (long)8 * (H + 2) * (W + 2) * 256;
    const long n16 = padElems * 2 / 16;
    {
      long blocks = (n16 + 255) / 256;
      if (blocks > 16384) blocks = 16384;
      fill_zero16<<<(unsigned)blocks, 256, 0, stream>>>((uint4*)padA, n16);
      fill_zero16<<<(unsigned)blocks, 256, 0, stream>>>((uint4*)padB, n16);
    }
    {
      long n = (long)8 * IN_CH * H * W;
      nchw_to_padded_nhwc<<<(unsigned)((n + 255) / 256), 256, 0, stream>>>(
          feats[l], padA, H, W);
    }
    dim3 cg((unsigned)(H * W / 64), 1, 8);
    conv3x3_wmma_relu<<<cg, 256, 0, stream>>>(padA, wpk0, reg_b0, padB, H, W);
    conv3x3_wmma_relu<<<cg, 256, 0, stream>>>(padB, wpk1, reg_b1, padC, H, W); // reg tower -> padC
    conv3x3_wmma_relu<<<cg, 256, 0, stream>>>(padA, wpk2, cls_b0, padB, H, W);
    conv3x3_wmma_relu<<<cg, 256, 0, stream>>>(padB, wpk3, cls_b1, padA, H, W); // cls tower -> padA
    {
      long n = (long)8 * H * W * NUM_ANCHORS;
      head_epilogue<<<(unsigned)((n + 255) / 256), 256, 0, stream>>>(
          padC, padA, reg_head_w, reg_head_b, cls_head_w, cls_head_b,
          conf_w, conf_b, out, H, W, strides[l], lvlOff[l]);
    }
  }
}
